// WindowedSelfAttention_64991445123113
// MI455X (gfx1250) — matrix operand order repaired and checked by reading, not compiled, not run on hardware
//
#include <hip/hip_runtime.h>
#include <hip/hip_bf16.h>
#include <math.h>

typedef _Float16 v16h __attribute__((ext_vector_type(16)));
typedef _Float16 v8h  __attribute__((ext_vector_type(8)));
typedef _Float16 h4   __attribute__((ext_vector_type(4)));
typedef _Float16 h2   __attribute__((ext_vector_type(2)));
typedef float    v8f  __attribute__((ext_vector_type(8)));
typedef float    f4   __attribute__((ext_vector_type(4)));

#define S_LEN 4096
#define DIM   128
#define HALFW 64
#define QT    64            // queries per workgroup (4 waves x 16)
#define KSPAN 192           // QT + 2*HALFW keys staged per workgroup
#define WSPAN 144           // per-wave key span (16 + 128), 9 tiles of 16
#define PSPAN 160           // P buffer padded to 5 x 32 keys
#define VPAD  208           // padded key stride of transposed V (32B-aligned rows)

__global__ __launch_bounds__(128)
void WindowedSelfAttention_64991445123113_kernel(const float* __restrict__ Qg,
                                                 const float* __restrict__ Kg,
                                                 const float* __restrict__ Vg,
                                                 float* __restrict__ Og)
{
    __shared__ __attribute__((aligned(32))) _Float16 sK[KSPAN * DIM];     // keys row-major f16
    __shared__ __attribute__((aligned(32))) _Float16 sVt[DIM * VPAD];     // values transposed [dim][key]
    __shared__ float    sS[4][16 * WSPAN];                                // masked scaled scores
    __shared__ __attribute__((aligned(8))) _Float16 sP[4][16 * PSPAN];    // exp(scores - max), f16
    __shared__ float    sTot[4][16];                                      // softmax denominators

    const int tid  = threadIdx.x;
    const int wave = tid >> 5;
    const int lane = tid & 31;
    const int hi   = lane >> 4;    // half-wave select
    const int ln   = lane & 15;

    const int blocksPerBatch = S_LEN / QT;
    const int b  = blockIdx.x / blocksPerBatch;
    const int q0 = (blockIdx.x % blocksPerBatch) * QT;

    const float* Qb = Qg + (size_t)b * S_LEN * DIM;
    const float* Kb = Kg + (size_t)b * S_LEN * DIM;
    const float* Vb = Vg + (size_t)b * S_LEN * DIM;
    float*       Ob = Og + (size_t)b * S_LEN * DIM;

    // ---------------- cooperative stage: K (f16) and V (f16, transposed) ----------------
    const int kbase = q0 - HALFW;              // absolute key index of LDS column 0
    for (int idx = tid; idx < KSPAN * (DIM / 4); idx += 128) {
        const int row = idx >> 5;              // key (workgroup-relative)
        const int c4  = idx & 31;              // float4 chunk within the 128-dim row
        const int j   = kbase + row;
        f4 kv = {}; f4 vv = {};
        if (j >= 0 && j < S_LEN) {
            kv = *(const f4*)(Kb + (size_t)j * DIM + 4 * c4);
            vv = *(const f4*)(Vb + (size_t)j * DIM + 4 * c4);
        }
        h4 kh;
        kh.x = (_Float16)kv.x; kh.y = (_Float16)kv.y;
        kh.z = (_Float16)kv.z; kh.w = (_Float16)kv.w;
        *(h4*)&sK[row * DIM + 4 * c4] = kh;
        sVt[(4 * c4 + 0) * VPAD + row] = (_Float16)vv.x;
        sVt[(4 * c4 + 1) * VPAD + row] = (_Float16)vv.y;
        sVt[(4 * c4 + 2) * VPAD + row] = (_Float16)vv.z;
        sVt[(4 * c4 + 3) * VPAD + row] = (_Float16)vv.w;
    }
    __syncthreads();

    // ---------------- Q fragments in WMMA A-layout, loaded once from global ----------------
    // A 16x32 f16 layout: lane row = lane%16; VGPR v holds K pair {kin, kin+1},
    // kin = 2v + (lane>=16 ? 16 : 0)   (contiguous k order, matches B)
    const int qrow = q0 + 16 * wave + ln;
    v16h qf[4];
#pragma unroll
    for (int kk = 0; kk < 4; ++kk) {
#pragma unroll
        for (int v = 0; v < 8; ++v) {
            const int kin = 2 * v + (hi ? 16 : 0);
            const float* p = Qb + (size_t)qrow * DIM + 32 * kk + kin;
            qf[kk][2 * v + 0] = (_Float16)p[0];
            qf[kk][2 * v + 1] = (_Float16)p[1];
        }
    }

    const int wkb  = q0 + 16 * wave - HALFW;   // abs key of wave-relative column 0
    const int wrel = 16 * wave;                // wave column 0 within the workgroup stage
    const float scale = 0.08838834764831845f;  // 1/sqrt(128)

    // ---------------- scores S = Q K^T over 9 key tiles ----------------
    for (int kt = 0; kt < 9; ++kt) {
        v8f c = {};
#pragma unroll
        for (int kk = 0; kk < 4; ++kk) {
            // B 32x16: lane holds key column (wrel+16*kt+ln); dims 32*kk + 16*hi + 0..15
            const _Float16* src = &sK[(wrel + 16 * kt + ln) * DIM + 32 * kk + 16 * hi];
            union { v16h v; v8h h[2]; } bf;
            bf.h[0] = *(const v8h*)(src);
            bf.h[1] = *(const v8h*)(src + 8);
            c = __builtin_amdgcn_wmma_f32_16x16x32_f16(false, qf[kk], false, bf.v,
                                                       (short)0, c, false, false);
        }
#pragma unroll
        for (int i = 0; i < 8; ++i) {
            const int m   = i + 8 * hi;             // query row in tile
            const int col = 16 * kt + ln;           // wave-relative key column
            const int j   = wkb + col;              // absolute key
            const bool valid = (col >= m) && (col <= m + 2 * HALFW) && (j >= 0) && (j < S_LEN);
            sS[wave][m * WSPAN + col] = valid ? c[i] * scale : -1e30f;
        }
    }
    asm volatile("s_wait_dscnt 0x0" ::: "memory");   // wave-local LDS RAW fence

    // ---------------- softmax (each lane: half of one row) ----------------
    {
        const int r  = ln;
        const int c0 = hi * (WSPAN / 2);
        float mx = -1e30f;
        for (int cc = 0; cc < WSPAN / 2; ++cc)
            mx = fmaxf(mx, sS[wave][r * WSPAN + c0 + cc]);
        mx = fmaxf(mx, __shfl_xor(mx, 16, 32));
        float sum = 0.f;
        for (int cc = 0; cc < WSPAN / 2; ++cc) {
            const float e = __expf(sS[wave][r * WSPAN + c0 + cc] - mx);
            sum += e;
            sP[wave][r * PSPAN + c0 + cc] = (_Float16)e;
        }
        sum += __shfl_xor(sum, 16, 32);
        if (hi) {   // zero the 144..159 pad so the last PV K-chunk is inert
            for (int cc = WSPAN; cc < PSPAN; ++cc)
                sP[wave][r * PSPAN + cc] = (_Float16)0.f;
        } else {
            sTot[wave][r] = sum;
        }
    }
    asm volatile("s_wait_dscnt 0x0" ::: "memory");

    float rden[8];
#pragma unroll
    for (int i = 0; i < 8; ++i)
        rden[i] = 1.0f / sTot[wave][i + 8 * hi];

    // ---------------- O = (P V) / denom over 8 head-dim tiles ----------------
    for (int nt = 0; nt < 8; ++nt) {
        v8f o = {};
#pragma unroll
        for (int kp = 0; kp < 5; ++kp) {
            // A = P fragment (row ln, key pairs per A-layout)
            v16h af;
#pragma unroll
            for (int v = 0; v < 8; ++v) {
                const int kin = 2 * v + (hi ? 16 : 0);
                h2 pv = *(const h2*)&sP[wave][ln * PSPAN + 32 * kp + kin];
                af[2 * v + 0] = pv.x;
                af[2 * v + 1] = pv.y;
            }
            // B = V fragment: column dim = nt*16+ln; keys 32*kp + 16*hi + 0..15 (contiguous in sVt)
            const _Float16* src = &sVt[(nt * 16 + ln) * VPAD + wrel + 32 * kp + 16 * hi];
            union { v16h v; v8h h[2]; } bf;
            bf.h[0] = *(const v8h*)(src);
            bf.h[1] = *(const v8h*)(src + 8);
            o = __builtin_amdgcn_wmma_f32_16x16x32_f16(false, af, false, bf.v,
                                                       (short)0, o, false, false);
        }
#pragma unroll
        for (int i = 0; i < 8; ++i) {
            const int m = i + 8 * hi;
            Ob[(size_t)(q0 + 16 * wave + m) * DIM + nt * 16 + ln] = o[i] * rden[i];
        }
    }
}

extern "C" void kernel_launch(void* const* d_in, const int* in_sizes, int n_in,
                              void* d_out, int out_size, void* d_ws, size_t ws_size,
                              hipStream_t stream) {
    const float* q = (const float*)d_in[0];
    const float* k = (const float*)d_in[1];
    const float* v = (const float*)d_in[2];
    float* o = (float*)d_out;

    int B = in_sizes[0] / (S_LEN * DIM);
    if (B < 1) B = 1;
    dim3 grid(B * (S_LEN / QT));
    WindowedSelfAttention_64991445123113_kernel<<<grid, dim3(128), 0, stream>>>(q, k, v, o);
}
